// Attention_47236050322078
// MI455X (gfx1250) — matrix operand order repaired and checked by reading, not compiled, not run on hardware
//
#include <hip/hip_runtime.h>

#define BB 4
#define SEQ 2048
#define HID 1024
#define NHEAD 16
#define DHEAD 64

// Flip to 0 if the gfx1250 assembler rejects the async mnemonics.
#define USE_ASYNC 1

typedef __attribute__((ext_vector_type(16))) __bf16          v16bf;
typedef __attribute__((ext_vector_type(8)))  float           v8f;
typedef __attribute__((ext_vector_type(16))) unsigned short  v16u16;
typedef __attribute__((ext_vector_type(8)))  unsigned short  v8u16;

__device__ __forceinline__ unsigned short f2bf(float f) {
  union { float f; unsigned u; } c; c.f = f;
  unsigned u = c.u;
  u += 0x7FFFu + ((u >> 16) & 1u);        // round-to-nearest-even
  return (unsigned short)(u >> 16);
}

// single-instruction max for finite inputs: med3(a, b, +inf) == max(a, b)
__device__ __forceinline__ float fmaxfast(float a, float b) {
  return __builtin_amdgcn_fmed3f(a, b, __builtin_inff());
}

// A fragment 16x32 bf16 from row-major [rows, ld] data.
// Contiguous k order used here: lane group g = lane/16, row M = lane%16, element e -> K = e + 16g
// => per lane two contiguous 16B chunks at k0+16g and k0+16g+8.
__device__ __forceinline__ v16bf frag_a_rowmajor(const unsigned short* base, int row, int ld, int k0) {
  const int lane = threadIdx.x & 31;
  const int g = lane >> 4;
  const unsigned short* p = base + (size_t)(row + (lane & 15)) * ld + k0 + 16 * g;
  v8u16 lo = *reinterpret_cast<const v8u16*>(p);
  v8u16 hi = *reinterpret_cast<const v8u16*>(p + 8);
  v16u16 u;
#pragma unroll
  for (int e = 0; e < 8; ++e) { u[e] = lo[e]; u[e + 8] = hi[e]; }
  return __builtin_bit_cast(v16bf, u);
}

// B fragment 32x16 bf16. Contiguous k order (same as A): N = lane%16, element e -> K = e + 16g.
// Data stored K-contiguous per N: one 32B load at base + (n0+N)*ldn + k0 + 16g.
__device__ __forceinline__ v16bf frag_b_kcontig(const unsigned short* base, int n0, size_t ldn, int k0) {
  const int lane = threadIdx.x & 31;
  const int g = lane >> 4;
  const unsigned short* p = base + (size_t)(n0 + (lane & 15)) * ldn + k0 + 16 * g;
  v16u16 u = *reinterpret_cast<const v16u16*>(p);
  return __builtin_bit_cast(v16bf, u);
}

__device__ __forceinline__ v8f wmma_bf16(v16bf a, v16bf b, v8f c) {
  return __builtin_amdgcn_wmma_f32_16x16x32_bf16(false, a, false, b, (short)0, c, false, false);
}

// Async global->LDS 16B copy (ASYNCcnt-tracked, no VGPR round trip).
__device__ __forceinline__ void async_b128(const void* gsrc, void* ldsdst) {
#if USE_ASYNC
  unsigned l = (unsigned)(unsigned long long)ldsdst;   // low 32 bits = LDS offset
  asm volatile("global_load_async_to_lds_b128 %0, %1, off"
               :: "v"(l), "v"(gsrc) : "memory");
#else
  *reinterpret_cast<uint4*>(ldsdst) = *reinterpret_cast<const uint4*>(gsrc);
#endif
}
__device__ __forceinline__ void wait_async_copies() {
#if USE_ASYNC
  asm volatile("s_wait_asynccnt 0x0" ::: "memory");
#endif
}

// ---------------------------------------------------------------------------
// Kernel 0: f32 -> bf16 bulk convert (coalesced, float4 -> 4x bf16)
// ---------------------------------------------------------------------------
__global__ __launch_bounds__(256) void cvt_k(const float4* __restrict__ s,
                                             uint2* __restrict__ d, int n4) {
  int i = blockIdx.x * 256 + threadIdx.x;
  if (i >= n4) return;
  float4 v = s[i];
  unsigned lo = (unsigned)f2bf(v.x) | ((unsigned)f2bf(v.y) << 16);
  unsigned hi = (unsigned)f2bf(v.z) | ((unsigned)f2bf(v.w) << 16);
  d[i] = make_uint2(lo, hi);
}

// ---------------------------------------------------------------------------
// Unified bf16 GEMM: O[M,cols] = A[M,1024] @ W[cols,1024]^T
// block 256 (8 waves), tile 256(M) x 128(N), per-wave 32x128 (16 wmma/K-step),
// K-step 32, double-buffered async-LDS staging.
// EPI==0: QKV epilogue (scale Q, scatter Q/K row-major + V transposed, bf16)
// EPI==1: bias add, f32 output
// ---------------------------------------------------------------------------
template <int EPI>
__global__ __launch_bounds__(256) void gemm_bf16_k(
    const unsigned short* __restrict__ A,   // [M,1024] bf16
    const unsigned short* __restrict__ W,   // [cols,1024] bf16
    unsigned short* __restrict__ qw, unsigned short* __restrict__ kw,
    unsigned short* __restrict__ vt,
    const float* __restrict__ bias, float* __restrict__ out) {
  __shared__ __align__(32) unsigned short As[2][256 * 32];   // 2 x 16KB
  __shared__ __align__(32) unsigned short Bs[2][128 * 32];   // 2 x 8KB
  const int tid = threadIdx.x;
  const int lane = tid & 31, wid = tid >> 5;
  const int g = lane >> 4, ln = lane & 15;
  const int cBase = blockIdx.x * 128;
  const int mBase = blockIdx.y * 256;

  auto stage = [&](int buf, int kb) {
#pragma unroll
    for (int t = 0; t < 4; ++t) {           // A tile 256x32: 1024 x b128
      int i = tid + t * 256;
      int row = i >> 2, c8 = (i & 3) << 3;
      async_b128(A + (size_t)(mBase + row) * HID + kb + c8, &As[buf][row * 32 + c8]);
    }
#pragma unroll
    for (int t = 0; t < 2; ++t) {           // B tile 128x32: 512 x b128
      int i = tid + t * 256;
      int n = i >> 2, c8 = (i & 3) << 3;
      async_b128(W + (size_t)(cBase + n) * HID + kb + c8, &Bs[buf][n * 32 + c8]);
    }
  };

  v8f acc[2][8] = {};
  stage(0, 0);
  int cur = 0;
  for (int kb = 0; kb < HID; kb += 32) {
    wait_async_copies();
    __syncthreads();                         // buffer `cur` ready for all waves
    if (kb + 32 < HID) stage(cur ^ 1, kb + 32);

    v16bf a0 = frag_a_rowmajor(&As[cur][0], wid * 32, 32, 0);
    v16bf a1 = frag_a_rowmajor(&As[cur][0], wid * 32 + 16, 32, 0);
#pragma unroll
    for (int f = 0; f < 8; ++f) {
      v16bf b = frag_b_kcontig(&Bs[cur][0], f * 16, 32, 0);
      acc[0][f] = wmma_bf16(a0, b, acc[0][f]);
      acc[1][f] = wmma_bf16(a1, b, acc[1][f]);
    }
    __syncthreads();                         // all reads of `cur` done before reuse
    cur ^= 1;
  }

#pragma unroll
  for (int r = 0; r < 2; ++r) {
#pragma unroll
    for (int f = 0; f < 8; ++f) {
      int col = cBase + f * 16 + ln;
      if constexpr (EPI == 0) {
        int mat = col >> 10, rem = col & 1023;
        int h = rem >> 6, d = rem & 63;
#pragma unroll
        for (int v = 0; v < 8; ++v) {
          int tok = mBase + wid * 32 + r * 16 + 8 * g + v;
          int b = tok >> 11, n = tok & (SEQ - 1);
          float val = acc[r][f][v];
          if (mat == 0) val *= 0.125f;       // DH^-0.5
          unsigned short bv = f2bf(val);
          if (mat == 2)
            vt[(((size_t)b * NHEAD + h) * DHEAD + d) * SEQ + n] = bv;
          else {
            unsigned short* dst = (mat == 0) ? qw : kw;
            dst[(((size_t)b * NHEAD + h) * SEQ + n) * DHEAD + d] = bv;
          }
        }
      } else {
        float bb = bias[col];
#pragma unroll
        for (int v = 0; v < 8; ++v) {
          int tok = mBase + wid * 32 + r * 16 + 8 * g + v;
          out[(size_t)tok * HID + col] = acc[r][f][v] + bb;
        }
      }
    }
  }
}

// ---------------------------------------------------------------------------
// Flash attention: block = 4 waves = 128 query rows of one head.
// K (32x64) and V^T (64x32) tiles async-staged to LDS, double-buffered, shared
// by all waves. Row sums of P computed on the matrix engine (P x ones).
// ---------------------------------------------------------------------------
__global__ __launch_bounds__(128) void flash_attn_k(const unsigned short* __restrict__ qw,
                                                    const unsigned short* __restrict__ kw,
                                                    const unsigned short* __restrict__ vt,
                                                    unsigned short* __restrict__ attn) {
  __shared__ __align__(32) unsigned short Ks[2][32 * 64];    // 2 x 4KB
  __shared__ __align__(32) unsigned short Vs[2][64 * 32];    // 2 x 4KB
  __shared__ __align__(32) unsigned short Pbuf[8][16 * 32];  // per (wave, r)
  const int tid = threadIdx.x, lane = tid & 31, wid = tid >> 5;
  const int g = lane >> 4, ln = lane & 15;
  const int bh = blockIdx.y;
  const int q0 = blockIdx.x * 128 + wid * 32;
  const unsigned short* qh = qw + (size_t)bh * SEQ * DHEAD;
  const unsigned short* kh = kw + (size_t)bh * SEQ * DHEAD;
  const unsigned short* vh = vt + (size_t)bh * DHEAD * SEQ;

  auto stageKV = [&](int buf, int kb) {
#pragma unroll
    for (int t = 0; t < 2; ++t) {            // K tile 32x64: 256 x b128
      int i = tid + t * 128;
      int row = i >> 3, c8 = (i & 7) << 3;
      async_b128(kh + (size_t)(kb + row) * DHEAD + c8, &Ks[buf][row * 64 + c8]);
    }
#pragma unroll
    for (int t = 0; t < 2; ++t) {            // V^T tile 64x32: 256 x b128
      int i = tid + t * 128;
      int d = i >> 2, c8 = (i & 3) << 3;
      async_b128(vh + (size_t)d * SEQ + kb + c8, &Vs[buf][d * 32 + c8]);
    }
  };

  // constant all-ones B fragment (bf16 1.0 = 0x3F80) for matrix-engine row sums
  v16u16 onesu;
#pragma unroll
  for (int e = 0; e < 16; ++e) onesu[e] = 0x3F80;
  const v16bf bones = __builtin_bit_cast(v16bf, onesu);

  float m[2][8], l[2][8];
  v8f o[2][4] = {};
#pragma unroll
  for (int r = 0; r < 2; ++r)
#pragma unroll
    for (int v = 0; v < 8; ++v) { m[r][v] = -1e30f; l[r][v] = 0.f; }

  v16bf aq[2][2];
#pragma unroll
  for (int r = 0; r < 2; ++r) {
    aq[r][0] = frag_a_rowmajor(qh, q0 + r * 16, DHEAD, 0);
    aq[r][1] = frag_a_rowmajor(qh, q0 + r * 16, DHEAD, 32);
  }

  stageKV(0, 0);
  int cur = 0;
  for (int kb = 0; kb < SEQ; kb += 32) {
    wait_async_copies();
    __syncthreads();                          // K/V tiles ready for all waves
    if (kb + 32 < SEQ) stageKV(cur ^ 1, kb + 32);

    v16bf bk0a = frag_b_kcontig(&Ks[cur][0], 0,  64, 0);
    v16bf bk0b = frag_b_kcontig(&Ks[cur][0], 0,  64, 32);
    v16bf bk1a = frag_b_kcontig(&Ks[cur][0], 16, 64, 0);
    v16bf bk1b = frag_b_kcontig(&Ks[cur][0], 16, 64, 32);
    v16bf bv[4];
#pragma unroll
    for (int f = 0; f < 4; ++f) bv[f] = frag_b_kcontig(&Vs[cur][0], f * 16, 32, 0);

#pragma unroll
    for (int r = 0; r < 2; ++r) {
      v8f s0 = {}, s1 = {};
      s0 = wmma_bf16(aq[r][0], bk0a, s0);
      s0 = wmma_bf16(aq[r][1], bk0b, s0);
      s1 = wmma_bf16(aq[r][0], bk1a, s1);
      s1 = wmma_bf16(aq[r][1], bk1b, s1);

      float alpha[8];
#pragma unroll
      for (int v = 0; v < 8; ++v) {
        float t = fmaxfast(s0[v], s1[v]);
#pragma unroll
        for (int off = 8; off; off >>= 1) t = fmaxfast(t, __shfl_xor(t, off, 32));
        float mn = fmaxfast(m[r][v], t);
        float p0 = __expf(s0[v] - mn);
        float p1 = __expf(s1[v] - mn);
        alpha[v] = __expf(m[r][v] - mn);
        m[r][v] = mn;
        Pbuf[wid * 2 + r][(v + 8 * g) * 32 + ln]      = f2bf(p0);
        Pbuf[wid * 2 + r][(v + 8 * g) * 32 + ln + 16] = f2bf(p1);
      }
#pragma unroll
      for (int f = 0; f < 4; ++f)
#pragma unroll
        for (int v = 0; v < 8; ++v) o[r][f][v] *= alpha[v];

      // same-wave DS ops are in-order (DScnt): no barrier needed for own Pbuf
      v16bf ap = frag_a_rowmajor(&Pbuf[wid * 2 + r][0], 0, 32, 0);

      // row sums of P on the matrix engine: rs[v] = sum_j P[row, j]
      v8f zz = {};
      v8f rs = wmma_bf16(ap, bones, zz);
#pragma unroll
      for (int v = 0; v < 8; ++v) l[r][v] = l[r][v] * alpha[v] + rs[v];

#pragma unroll
      for (int f = 0; f < 4; ++f) o[r][f] = wmma_bf16(ap, bv[f], o[r][f]);
    }
    __syncthreads();                          // all waves done with cur tiles
    cur ^= 1;
  }

  const int b = bh >> 4, h = bh & 15;
#pragma unroll
  for (int r = 0; r < 2; ++r)
#pragma unroll
    for (int f = 0; f < 4; ++f) {
      int d = f * 16 + ln;
#pragma unroll
      for (int v = 0; v < 8; ++v) {
        int nq = q0 + r * 16 + v + 8 * g;
        attn[(size_t)(b * SEQ + nq) * HID + h * DHEAD + d] = f2bf(o[r][f][v] / l[r][v]);
      }
    }
}

// ---------------------------------------------------------------------------
extern "C" void kernel_launch(void* const* d_in, const int* in_sizes, int n_in,
                              void* d_out, int out_size, void* d_ws, size_t ws_size,
                              hipStream_t stream) {
  const float* x     = (const float*)d_in[0];
  const float* w_qkv = (const float*)d_in[1];
  const float* w_out = (const float*)d_in[2];
  const float* b_out = (const float*)d_in[3];
  float* out = (float*)d_out;

  unsigned short* ws = (unsigned short*)d_ws;
  const size_t per = (size_t)BB * NHEAD * SEQ * DHEAD;   // 8M bf16 elements
  unsigned short* xb    = ws;                            // x as bf16 (reused as attn later)
  unsigned short* qw    = ws + per;
  unsigned short* kw    = ws + 2 * per;
  unsigned short* vt    = ws + 3 * per;
  unsigned short* wqkvb = ws + 4 * per;                  // 3M elems
  unsigned short* woutb = wqkvb + (size_t)3 * HID * HID; // 1M elems -> total 72MB
  unsigned short* attn  = xb;                            // alias: qkv reads xb before flash writes

  {
    int n4x = BB * SEQ * HID / 4;
    int n4q = 3 * HID * HID / 4;
    int n4o = HID * HID / 4;
    cvt_k<<<dim3((n4x + 255) / 256), dim3(256), 0, stream>>>((const float4*)x, (uint2*)xb, n4x);
    cvt_k<<<dim3((n4q + 255) / 256), dim3(256), 0, stream>>>((const float4*)w_qkv, (uint2*)wqkvb, n4q);
    cvt_k<<<dim3((n4o + 255) / 256), dim3(256), 0, stream>>>((const float4*)w_out, (uint2*)woutb, n4o);
  }

  gemm_bf16_k<0><<<dim3(3 * HID / 128, BB * SEQ / 256), dim3(256), 0, stream>>>(
      xb, wqkvb, qw, kw, vt, nullptr, nullptr);
  flash_attn_k<<<dim3(SEQ / 128, BB * NHEAD), dim3(128), 0, stream>>>(qw, kw, vt, attn);
  gemm_bf16_k<1><<<dim3(HID / 128, BB * SEQ / 256), dim3(256), 0, stream>>>(
      attn, woutb, nullptr, nullptr, nullptr, b_out, out);
}
